// MultiHeadAttention_22591527977488
// MI455X (gfx1250) — matrix operand order repaired and checked by reading, not compiled, not run on hardware
//
#include <hip/hip_runtime.h>
#include <hip/hip_bf16.h>

#define DIN   1024
#define DOUT  64
#define NH    16
#define SEQ   2048
#define BATCH 4
#define KTILES (SEQ / 32)   // 64 key tiles per (b,h)
#define KSTEPS (DIN / 32)   // 32 k-steps in the projection
#define QWAVES 8            // waves per attention block (128 q rows)

typedef __attribute__((ext_vector_type(16))) __bf16 v16bf;
typedef __attribute__((ext_vector_type(8)))  __bf16 v8bf;
typedef __attribute__((ext_vector_type(8)))  float  v8f;
typedef __attribute__((ext_vector_type(4)))  float  v4f;
typedef __attribute__((ext_vector_type(4)))  __bf16 v4bf;
typedef __attribute__((ext_vector_type(4)))  unsigned uint4v;
typedef __attribute__((ext_vector_type(8)))  int     int8v;
typedef __attribute__((ext_vector_type(4)))  int     int4v;

#if defined(__HIP_DEVICE_COMPILE__) && __has_builtin(__builtin_amdgcn_tensor_load_to_lds) && \
    __has_builtin(__builtin_amdgcn_s_wait_tensorcnt)
#define USE_TDM 1
#else
#define USE_TDM 0
#endif

// ---------------- wave32 cross-lane helpers (ds_bpermute butterflies) ----------
__device__ __forceinline__ float xor_lane(float v, int mask) {
  int src = (((int)threadIdx.x & 31) ^ mask) << 2;
  return __int_as_float(__builtin_amdgcn_ds_bpermute(src, __float_as_int(v)));
}
__device__ __forceinline__ float half_max16(float v) {   // reduce across 16-lane half
  v = fmaxf(v, xor_lane(v, 1));
  v = fmaxf(v, xor_lane(v, 2));
  v = fmaxf(v, xor_lane(v, 4));
  v = fmaxf(v, xor_lane(v, 8));
  return v;
}
__device__ __forceinline__ float half_sum16(float v) {
  v += xor_lane(v, 1);
  v += xor_lane(v, 2);
  v += xor_lane(v, 4);
  v += xor_lane(v, 8);
  return v;
}

// ---------------- WMMA fragment loaders (layouts per CDNA5 ISA) ----
// A fragment: 16x32 bf16 (MxK) from row-major [M][K]; per-lane contiguous 16B
// chunks -> compiler emits 2x b128 loads (verified in earlier rounds' asm).
__device__ __forceinline__ v16bf load_a16x32(const __bf16* base, int ld, int lane) {
  v16bf a;
  const int M  = lane & 15;
  const int hi = (lane >> 4) & 1;
#pragma unroll
  for (int i = 0; i < 8; ++i) {
    const int kb = (hi << 4) + (i << 1);
    union { unsigned u; __bf16 h[2]; } t;
    t.u = *(const unsigned*)(base + (size_t)M * ld + kb);
    a[2 * i]     = t.h[0];
    a[2 * i + 1] = t.h[1];
  }
  return a;
}

// B fragment from a pre-swizzled 1KB block: lane=(N)|(khalf bit16), j=K in half.
__device__ __forceinline__ v16bf load_bfrag_packed(const __bf16* blk, int lane) {
  return *(const v16bf*)(blk + (size_t)lane * 16);
}

// ---------------- TDM: 1-D tile load Global -> LDS (D# per CDNA5 ISA) -
#if USE_TDM
__device__ __forceinline__ void tdm_load_1d(unsigned lds_addr, const void* gaddr,
                                            unsigned nelem /* 2-byte elems */) {
  const unsigned long long ga = (unsigned long long)(uintptr_t)gaddr;
  uint4v g0;
  g0[0] = 1u;                                   // count=1 (valid user D#)
  g0[1] = lds_addr;                             // lds_addr (bytes)
  g0[2] = (unsigned)ga;                         // global_addr[31:0]
  g0[3] = (unsigned)((ga >> 32) & 0x01FFFFFFu)  // global_addr[56:32]
        | (2u << 30);                           // type=2 ("image")
  int8v g1;
  g1[0] = (int)(1u << 16);                      // data_size=1 -> 2 bytes
  g1[1] = (int)(nelem << 16);                   // tensor_dim0[15:0]
  g1[2] = (int)(1u << 16);                      // tensor_dim0 hi=0, tensor_dim1=1
  g1[3] = (int)(nelem << 16);                   // tile_dim0 = nelem
  g1[4] = 1;                                    // tile_dim1=1, tile_dim2=0
  g1[5] = (int)nelem;                           // tensor_dim0_stride
  g1[6] = 0;
  g1[7] = 0;
  int4v z4 = {0, 0, 0, 0};
#if defined(__clang_major__) && (__clang_major__ >= 23)
  int8v z8 = {0, 0, 0, 0, 0, 0, 0, 0};
  __builtin_amdgcn_tensor_load_to_lds(g0, g1, z4, z4, z8, 0);
#else
  __builtin_amdgcn_tensor_load_to_lds(g0, g1, z4, z4, 0);
#endif
}
#endif

// ---------------- f32 -> bf16 conversion (vectorized) --------------------------
__global__ __launch_bounds__(256) void cvt_f32_bf16(const v4f* __restrict__ src,
                                                    v4bf* __restrict__ dst, int n4) {
  int i = blockIdx.x * blockDim.x + threadIdx.x;
  if (i < n4) {
    v4f x = src[i];
    v4bf y;
    y[0] = (__bf16)x[0]; y[1] = (__bf16)x[1];
    y[2] = (__bf16)x[2]; y[3] = (__bf16)x[3];
    dst[i] = y;
  }
}

// ---------------- pack W[h, k, n] (f32) into B-fragment-major bf16 blocks ------
__global__ __launch_bounds__(256) void pack_w_swz(const float* __restrict__ src,
                                                  __bf16* __restrict__ dst, int n) {
  int idx = blockIdx.x * blockDim.x + threadIdx.x;
  if (idx >= n) return;
  const int h   = idx / (DIN * DOUT);
  const int rem = idx - h * (DIN * DOUT);
  const int k   = rem / DOUT;
  const int e   = rem - k * DOUT;
  const int ks   = k >> 5;
  const int lane = (e & 15) | (k & 16);
  const int j    = k & 15;
  const int t    = e >> 4;
  dst[((((size_t)h * KSTEPS + ks) * 4 + t) << 9) + lane * 16 + j] = (__bf16)src[idx];
}

// ---------------- QKV projection: one wave computes a 16x64 tile ---------------
__global__ __launch_bounds__(32) void qkv_proj(
    const __bf16* __restrict__ Xb,
    const __bf16* __restrict__ WqS, const __bf16* __restrict__ WkS,
    const __bf16* __restrict__ WvS,
    __bf16* __restrict__ Qb,                              // [B,NH,S,DOUT] (x 1/8)
    __bf16* __restrict__ KtS,                             // score-B-fragment swizzled
    __bf16* __restrict__ VS) {                            // ctx-B-fragment swizzled
  const int lane = threadIdx.x & 31;
  const int row0 = blockIdx.x * 16;
  const int h    = blockIdx.y;
  const int mat  = blockIdx.z;

  const __bf16* W = (mat == 0) ? WqS : (mat == 1) ? WkS : WvS;
  W += ((size_t)h * KSTEPS * 4) << 9;
  const __bf16* X = Xb + (size_t)row0 * DIN;

  v8f acc[4] = {};
  for (int ks = 0; ks < KSTEPS; ++ks) {
    v16bf a = load_a16x32(X + ks * 32, DIN, lane);
    const __bf16* wblk = W + ((size_t)ks << 11);
#pragma unroll
    for (int t = 0; t < 4; ++t) {
      v16bf b = load_bfrag_packed(wblk + ((size_t)t << 9), lane);
      acc[t] = __builtin_amdgcn_wmma_f32_16x16x32_bf16(
          false, a, false, b, (short)0, acc[t], false, false);
    }
  }

  const int bidx = row0 / SEQ;
  const int s0   = row0 % SEQ;
  const int N    = lane & 15;
  const int hi8  = ((lane >> 4) & 1) << 3;
  const size_t bh = (size_t)bidx * NH + h;
#pragma unroll
  for (int t = 0; t < 4; ++t) {
#pragma unroll
    for (int r = 0; r < 8; ++r) {
      const int s = s0 + r + hi8;                         // sequence pos (= key)
      const int e = t * 16 + N;
      const float v = acc[t][r];
      if (mat == 0) {
        Qb[(bh * SEQ + s) * DOUT + e] = (__bf16)(v * 0.125f);   // fold 1/sqrt(64)
      } else if (mat == 1) {
        // score B frag: rows=e (K-dim), cols=key; lane=(key&15)|(e&16), j=e&15
        const size_t blk = ((bh * KTILES + (s >> 5)) * 2 + ((s >> 4) & 1)) * 2 + (e >> 5);
        KtS[(blk << 9) + ((s & 15) | (e & 16)) * 16 + (e & 15)] = (__bf16)v;
      } else {
        // ctx B frag: rows=key (K-dim), cols=e; lane=(e&15)|(key&16), j=key&15
        const size_t blk = (bh * KTILES + (s >> 5)) * 4 + (e >> 4);
        VS[(blk << 9) + ((e & 15) | (s & 16)) * 16 + (s & 15)] = (__bf16)v;
      }
    }
  }
}

// ---------------- causal flash attention ---------------------------------------
// block = 8 waves = 128 q rows of one (b,h); K/V tiles staged to LDS once per
// block (8x L2 traffic reduction) via TDM, double-buffered against compute.
// grid = (S/128, NH, B), block = 256
__global__ __launch_bounds__(256) void attn_fwd(
    const __bf16* __restrict__ Qb, const __bf16* __restrict__ KtS,
    const __bf16* __restrict__ VS, float* __restrict__ out) {
  __shared__ __align__(32) __bf16 kstage[2][2048];        // 4KB per key-tile
  __shared__ __align__(32) __bf16 vstage[2][2048];
  __shared__ __align__(32) __bf16 plds[QWAVES][512];      // per-wave P relayout

  const int tid   = threadIdx.x;
  const int lane  = tid & 31;
  const int wv    = tid >> 5;
  const int qbase = blockIdx.x * (16 * QWAVES) + wv * 16;
  const int h     = blockIdx.y;
  const int bb    = blockIdx.z;
  const size_t bh = (size_t)bb * NH + h;

  const __bf16* Qh = Qb  + bh * SEQ * DOUT;
  const __bf16* Kh = KtS + ((bh * KTILES * 4) << 9);      // 2048 elems per tile
  const __bf16* Vh = VS  + ((bh * KTILES * 4) << 9);

  const v16bf qa0 = load_a16x32(Qh + (size_t)qbase * DOUT,      DOUT, lane);
  const v16bf qa1 = load_a16x32(Qh + (size_t)qbase * DOUT + 32, DOUT, lane);

  v8f ctx[4] = {};
  float mrow[8], lrow[8];
#pragma unroll
  for (int r = 0; r < 8; ++r) { mrow[r] = -3.0e38f; lrow[r] = 0.0f; }

  const int nlo = lane & 15;
  const int hi8 = ((lane >> 4) & 1) << 3;
  const int mysteps    = (qbase + 16 + 31) >> 5;          // this wave's causal bound
  const int blocksteps = blockIdx.x * 4 + 4;              // last wave's bound

  // ---- stage tile 0
#if USE_TDM
  if (wv == 0) {
    tdm_load_1d((unsigned)(uintptr_t)&kstage[0][0], Kh, 2048);
    tdm_load_1d((unsigned)(uintptr_t)&vstage[0][0], Vh, 2048);
    __builtin_amdgcn_s_wait_tensorcnt(0);
  }
#else
  ((v8bf*)kstage[0])[tid] = ((const v8bf*)Kh)[tid];
  ((v8bf*)vstage[0])[tid] = ((const v8bf*)Vh)[tid];
#endif
  __syncthreads();

  for (int kt = 0; kt < blocksteps; ++kt) {
    const int buf = kt & 1;
    // ---- kick off next tile into the other buffer (async vs this compute)
    if (kt + 1 < blocksteps) {
#if USE_TDM
      if (wv == 0) {
        tdm_load_1d((unsigned)(uintptr_t)&kstage[buf ^ 1][0],
                    Kh + ((size_t)(kt + 1) << 11), 2048);
        tdm_load_1d((unsigned)(uintptr_t)&vstage[buf ^ 1][0],
                    Vh + ((size_t)(kt + 1) << 11), 2048);
      }
#else
      ((v8bf*)kstage[buf ^ 1])[tid] = ((const v8bf*)(Kh + ((size_t)(kt + 1) << 11)))[tid];
      ((v8bf*)vstage[buf ^ 1])[tid] = ((const v8bf*)(Vh + ((size_t)(kt + 1) << 11)))[tid];
#endif
    }

    if (kt < mysteps) {                                   // wave-uniform branch
      const __bf16* kblk = kstage[buf];
      const __bf16* vblk = vstage[buf];

      // ---- scores: S = Q * K^T  (two 16x16 C tiles)
      v8f s0 = {}, s1 = {};
      {
        v16bf b;
        b  = load_bfrag_packed(kblk + 0 * 512, lane);
        s0 = __builtin_amdgcn_wmma_f32_16x16x32_bf16(false, qa0, false, b, (short)0, s0, false, false);
        b  = load_bfrag_packed(kblk + 1 * 512, lane);
        s0 = __builtin_amdgcn_wmma_f32_16x16x32_bf16(false, qa1, false, b, (short)0, s0, false, false);
        b  = load_bfrag_packed(kblk + 2 * 512, lane);
        s1 = __builtin_amdgcn_wmma_f32_16x16x32_bf16(false, qa0, false, b, (short)0, s1, false, false);
        b  = load_bfrag_packed(kblk + 3 * 512, lane);
        s1 = __builtin_amdgcn_wmma_f32_16x16x32_bf16(false, qa1, false, b, (short)0, s1, false, false);
      }

      // ---- causal mask + online softmax
      const int k0 = kt * 32;
#pragma unroll
      for (int r = 0; r < 8; ++r) {
        const int q  = qbase + r + hi8;
        const float e0 = (k0 + nlo      <= q) ? s0[r] : -3.0e38f;
        const float e1 = (k0 + 16 + nlo <= q) ? s1[r] : -3.0e38f;
        const float mx    = half_max16(fmaxf(e0, e1));
        const float mnew  = fmaxf(mrow[r], mx);
        const float alpha = __expf(mrow[r] - mnew);
        const float p0    = __expf(e0 - mnew);
        const float p1    = __expf(e1 - mnew);
        lrow[r] = lrow[r] * alpha + half_sum16(p0 + p1);
        mrow[r] = mnew;
#pragma unroll
        for (int t = 0; t < 4; ++t) ctx[t][r] = ctx[t][r] * alpha;
        const int M = r + hi8;
        plds[wv][M * 32 + nlo]      = (__bf16)p0;         // C layout -> [M][key]
        plds[wv][M * 32 + 16 + nlo] = (__bf16)p1;
      }
      // same-wave LDS ops are in-order; compiler inserts the ds waits
      const v16bf pa = load_a16x32(plds[wv], 32, lane);

      // ---- ctx += P * V
#pragma unroll
      for (int t = 0; t < 4; ++t) {
        v16bf vb = load_bfrag_packed(vblk + ((size_t)t << 9), lane);
        ctx[t] = __builtin_amdgcn_wmma_f32_16x16x32_bf16(
            false, pa, false, vb, (short)0, ctx[t], false, false);
      }
    }

#if USE_TDM
    if (wv == 0 && kt + 1 < blocksteps) __builtin_amdgcn_s_wait_tensorcnt(0);
#endif
    __syncthreads();                                      // next buffer ready / safe
  }

  // ---- epilogue: out[b, s, h*64+e] = ctx / l
#pragma unroll
  for (int r = 0; r < 8; ++r) lrow[r] = 1.0f / lrow[r];
#pragma unroll
  for (int t = 0; t < 4; ++t) {
#pragma unroll
    for (int r = 0; r < 8; ++r) {
      const int s = qbase + r + hi8;
      const int e = t * 16 + nlo;
      out[((size_t)bb * SEQ + s) * (NH * DOUT) + h * DOUT + e] = ctx[t][r] * lrow[r];
    }
  }
}

// ---------------- host-side orchestration --------------------------------------
extern "C" void kernel_launch(void* const* d_in, const int* in_sizes, int n_in,
                              void* d_out, int out_size, void* d_ws, size_t ws_size,
                              hipStream_t stream) {
  (void)in_sizes; (void)n_in; (void)out_size; (void)ws_size;
  const float* X  = (const float*)d_in[0];
  const float* Wq = (const float*)d_in[1];
  const float* Wk = (const float*)d_in[2];
  const float* Wv = (const float*)d_in[3];
  float* out = (float*)d_out;

  char* ws = (char*)d_ws;
  size_t off = 0;
  auto take = [&](size_t elems) -> void* {
    void* p = ws + off;
    off += (elems * 2 + 255) & ~(size_t)255;
    return p;
  };
  const size_t nX = (size_t)BATCH * SEQ * DIN;
  const size_t nW = (size_t)NH * DIN * DOUT;
  const size_t nQ = (size_t)BATCH * NH * SEQ * DOUT;
  __bf16* Xb  = (__bf16*)take(nX);
  __bf16* WqS = (__bf16*)take(nW);
  __bf16* WkS = (__bf16*)take(nW);
  __bf16* WvS = (__bf16*)take(nW);
  __bf16* Qb  = (__bf16*)take(nQ);
  __bf16* KtS = (__bf16*)take(nQ);
  __bf16* VS  = (__bf16*)take(nQ);

  cvt_f32_bf16<<<(unsigned)((nX / 4 + 255) / 256), 256, 0, stream>>>(
      (const v4f*)X, (v4bf*)Xb, (int)(nX / 4));
  pack_w_swz<<<(unsigned)((nW + 255) / 256), 256, 0, stream>>>(Wq, WqS, (int)nW);
  pack_w_swz<<<(unsigned)((nW + 255) / 256), 256, 0, stream>>>(Wk, WkS, (int)nW);
  pack_w_swz<<<(unsigned)((nW + 255) / 256), 256, 0, stream>>>(Wv, WvS, (int)nW);

  qkv_proj<<<dim3((BATCH * SEQ) / 16, NH, 3), 32, 0, stream>>>(
      Xb, WqS, WkS, WvS, Qb, KtS, VS);

  attn_fwd<<<dim3(SEQ / (16 * QWAVES), NH, BATCH), 256, 0, stream>>>(Qb, KtS, VS, out);
}
